// _CEWithClassifier_249108103716
// MI455X (gfx1250) — matrix operand order repaired and checked by reading, not compiled, not run on hardware
//
#include <hip/hip_runtime.h>
#include <hip/hip_bf16.h>
#include <math.h>

#define EMBED_DIM   2048
#define NUM_CLASSES 50257
#define BATCH       2048
#define EPS         0.1f

#define BM 128                                   // classes per block (WMMA M)
#define BN 256                                   // batch rows per block (WMMA N)
#define BK 32                                    // K per iteration (one bf16 WMMA)
#define NBC ((NUM_CLASSES + BM - 1) / BM)        // 393 class tiles
#define NBB (BATCH / BN)                         // 8 batch tiles
#define NK  (EMBED_DIM / BK)                     // 64 K-steps

typedef __attribute__((ext_vector_type(16))) __bf16 bf16x16;
typedef __attribute__((ext_vector_type(8)))  __bf16 bf16x8;
typedef __attribute__((ext_vector_type(8)))  float  f32x8;

// Online-softmax merge with -inf guard
__device__ __forceinline__ void online_merge(float& M, float& S, float mo, float so)
{
    float Mn = fmaxf(M, mo);
    if (Mn == -INFINITY) { M = Mn; S = 0.0f; return; }
    S = S * expf(M - Mn) + so * expf(mo - Mn);
    M = Mn;
}

// ---------------------------------------------------------------------------
// Kernel 0: transpose + convert E [B, D] f32  ->  EtG [D, B] bf16 (once)
// ---------------------------------------------------------------------------
__global__ __launch_bounds__(256)
void e_transpose_bf16(const float* __restrict__ E, __bf16* __restrict__ EtG)
{
    __shared__ __bf16 tile[64][72];              // padded vs bank conflicts
    const int tid   = threadIdx.x;
    const int bBase = blockIdx.x * 64;
    const int dBase = blockIdx.y * 64;
    #pragma unroll
    for (int p = 0; p < 16; ++p) {
        int idx = p * 256 + tid;                 // 64x64 elements
        int br  = idx >> 6;
        int dc  = idx & 63;
        tile[dc][br] = (__bf16)E[(size_t)(bBase + br) * EMBED_DIM + dBase + dc];
    }
    __syncthreads();
    #pragma unroll
    for (int p = 0; p < 16; ++p) {
        int idx = p * 256 + tid;
        int dr  = idx >> 6;
        int bc  = idx & 63;
        EtG[(size_t)(dBase + dr) * BATCH + bBase + bc] = tile[dr][bc];
    }
}

// ---------------------------------------------------------------------------
// Epilogue: partial softmax stats over the block's full 128 classes.
// Each wave owns 32 batch rows; lane-half pair covers all 128 classes.
// MASK=true only for the final class tile (classes >= NUM_CLASSES exist).
// ---------------------------------------------------------------------------
template<bool MASK>
__device__ __forceinline__ void ce_epilogue(
    const f32x8 (&acc)[8][2],
    const int cBase, const int bBase, const int n0,
    const int lane, const int mAdd, const int cb,
    const int* __restrict__ labels,
    float* __restrict__ m_part, float* __restrict__ s_part,
    float* __restrict__ t_part, float* __restrict__ logit_y)
{
    #pragma unroll
    for (int ni = 0; ni < 2; ++ni) {
        const int rowL = n0 + ni * 16 + (lane & 15);
        const int rowG = bBase + rowL;
        float mx = -INFINITY, t = 0.0f, s = 0.0f;
        #pragma unroll
        for (int mi = 0; mi < 8; ++mi) {
            const int nval = NUM_CLASSES - (cBase + mi * 16 + mAdd);
            #pragma unroll
            for (int r = 0; r < 8; ++r) {
                const float x = acc[mi][ni][r];
                if (MASK) {
                    const bool v = r < nval;
                    mx = v ? fmaxf(mx, x) : mx;
                    t  = v ? t + x : t;
                } else {
                    mx = fmaxf(mx, x);
                    t += x;
                }
            }
        }
        #pragma unroll
        for (int mi = 0; mi < 8; ++mi) {
            const int nval = NUM_CLASSES - (cBase + mi * 16 + mAdd);
            #pragma unroll
            for (int r = 0; r < 8; ++r) {
                const float e = expf(acc[mi][ni][r] - mx);
                if (MASK) s = (r < nval) ? s + e : s;
                else      s += e;
            }
        }
        // branch-free target-class logit extraction
        const int lb = labels[rowG];
        float ly = 0.0f;
        bool found = false;
        #pragma unroll
        for (int mi = 0; mi < 8; ++mi) {
            const int d0 = lb - (cBase + mi * 16 + mAdd);
            found = found || ((unsigned)d0 < 8u);
            #pragma unroll
            for (int r = 0; r < 8; ++r)
                ly = (d0 == r) ? acc[mi][ni][r] : ly;
        }
        if (found) logit_y[rowG] = ly;

        // merge the two lane halves (classes r+0..7 with r+8..15)
        const float mo = __shfl_xor(mx, 16, 32);
        const float so = __shfl_xor(s, 16, 32);
        const float to = __shfl_xor(t, 16, 32);
        online_merge(mx, s, mo, so);
        t += to;
        if (lane < 16) {
            const size_t idx = (size_t)rowG * NBC + cb;
            m_part[idx] = mx;
            s_part[idx] = s;
            t_part[idx] = t;
        }
    }
}

// ---------------------------------------------------------------------------
// Kernel 1: bf16 WMMA GEMM tile + partial softmax stats
//   A = W tile [128 x 32], f32->bf16, double-buffered in LDS (reg staged)
//   B = EtG (K-major bf16 in global/L2; lane = batch col, element = K; 16 contiguous batch cols)
//   Per wave: 128 classes x 32 batch = 8x2 accumulators, 16 WMMA per K-step.
//   Unique B fragments per wave -> no L2 duplication across waves.
// ---------------------------------------------------------------------------
__global__ __launch_bounds__(256)
void ce_gemm_partials(const __bf16* __restrict__ EtG, const float* __restrict__ W,
                      const int* __restrict__ labels,
                      float* __restrict__ m_part, float* __restrict__ s_part,
                      float* __restrict__ t_part, float* __restrict__ logit_y)
{
    __shared__ __align__(32) __bf16 Wl[2][BM][BK];   // 16 KB double buffer (only LDS)

    const int tid   = threadIdx.x;
    const int lane  = tid & 31;
    const int wave  = tid >> 5;
    const int bBase = blockIdx.x * BN;   // batch fastest -> adjacent blocks share W in L2
    const int cBase = blockIdx.y * BM;

    const int n0   = wave * 32;          // each wave: its own 32 batch rows
    const int mAdd = (lane < 16) ? 0 : 8;

    // ---- W staging: register-buffered, double-buffered LDS ----
    const int sr  = tid >> 3;            // 0..31 (row within 32-row stripe)
    const int sc4 = tid & 7;             // float4 column
    const float* wptr[4];
    #pragma unroll
    for (int p = 0; p < 4; ++p) {
        int c = cBase + p * 32 + sr;
        if (c >= NUM_CLASSES) c = NUM_CLASSES - 1;   // clamp; masked in epilogue
        wptr[p] = W + (size_t)c * EMBED_DIM + sc4 * 4;
    }
    float4 stg[4];
    auto load_tile = [&](int kt) {
        #pragma unroll
        for (int p = 0; p < 4; ++p)
            stg[p] = *(const float4*)(wptr[p] + kt * BK);
    };
    auto store_tile = [&](int buf) {
        #pragma unroll
        for (int p = 0; p < 4; ++p) {
            union { __bf16 b[4]; uint2 u; } t;
            t.b[0] = (__bf16)stg[p].x; t.b[1] = (__bf16)stg[p].y;
            t.b[2] = (__bf16)stg[p].z; t.b[3] = (__bf16)stg[p].w;
            *(uint2*)&Wl[buf][p * 32 + sr][sc4 * 4] = t.u;
        }
    };

    const int arow = lane & 15;
    const int kOff = (lane < 16) ? 0 : 8;
    const __bf16* bptr = EtG + bBase + n0 + (lane & 15);

    f32x8 acc[8][2] = {};
    auto compute = [&](int buf, int kt) {
        const __bf16* brow = bptr + (size_t)kt * BK * BATCH;
        bf16x16 bfrag[2];
        #pragma unroll
        for (int ni = 0; ni < 2; ++ni)
            #pragma unroll
            for (int s = 0; s < 16; ++s)
                bfrag[ni][s] = brow[(size_t)((s < 8) ? kOff + s : kOff + 8 + s) * BATCH + ni * 16];
        #pragma unroll
        for (int mi = 0; mi < 8; ++mi) {
            const __bf16* rp = &Wl[buf][mi * 16 + arow][0];
            bf16x8 lo = *(const bf16x8*)(rp + kOff);
            bf16x8 hi = *(const bf16x8*)(rp + kOff + 16);
            const bf16x16 afrag = __builtin_shufflevector(lo, hi,
                0,1,2,3,4,5,6,7,8,9,10,11,12,13,14,15);
            acc[mi][0] = __builtin_amdgcn_wmma_f32_16x16x32_bf16(
                false, afrag, false, bfrag[0], (short)0, acc[mi][0], false, false);
            acc[mi][1] = __builtin_amdgcn_wmma_f32_16x16x32_bf16(
                false, afrag, false, bfrag[1], (short)0, acc[mi][1], false, false);
        }
    };

    load_tile(0);
    store_tile(0);
    for (int kt = 0; kt < NK - 1; ++kt) {
        __syncthreads();                 // buf[kt&1] ready; prior reads of other buf done
        load_tile(kt + 1);               // global loads overlap current compute
        compute(kt & 1, kt);
        store_tile((kt + 1) & 1);        // convert + stage next tile
    }
    __syncthreads();
    compute((NK - 1) & 1, NK - 1);

    // ---- partial softmax stats straight from accumulators to global ----
    if (cBase + BM <= NUM_CLASSES)
        ce_epilogue<false>(acc, cBase, bBase, n0, lane, mAdd, blockIdx.y,
                           labels, m_part, s_part, t_part, logit_y);
    else
        ce_epilogue<true>(acc, cBase, bBase, n0, lane, mAdd, blockIdx.y,
                          labels, m_part, s_part, t_part, logit_y);
}

// ---------------------------------------------------------------------------
// Kernel 2: one wave per batch row — merge 393 partials, emit per-row loss
// ---------------------------------------------------------------------------
__global__ __launch_bounds__(256)
void ce_row_reduce(const float* __restrict__ m_part, const float* __restrict__ s_part,
                   const float* __restrict__ t_part, const float* __restrict__ logit_y,
                   float* __restrict__ loss)
{
    const int lane = threadIdx.x & 31;
    const int wave = threadIdx.x >> 5;
    const int b    = blockIdx.x * 8 + wave;

    float M = -INFINITY, S = 0.0f, T = 0.0f;
    for (int j = lane; j < NBC; j += 32) {
        const size_t idx = (size_t)b * NBC + j;
        online_merge(M, S, m_part[idx], s_part[idx]);
        T += t_part[idx];
    }
    #pragma unroll
    for (int off = 16; off >= 1; off >>= 1) {
        const float mo = __shfl_xor(M, off, 32);
        const float so = __shfl_xor(S, off, 32);
        const float to = __shfl_xor(T, off, 32);
        online_merge(M, S, mo, so);
        T += to;
    }
    if (lane == 0) {
        const float lse = M + logf(S);
        loss[b] = lse - (1.0f - EPS) * logit_y[b] - EPS * (T / (float)NUM_CLASSES);
    }
}

// ---------------------------------------------------------------------------
// Kernel 3: mean over BATCH losses -> d_out[0]
// ---------------------------------------------------------------------------
__global__ __launch_bounds__(256)
void ce_final(const float* __restrict__ loss, float* __restrict__ out)
{
    __shared__ float red[256];
    float acc = 0.0f;
    for (int i = threadIdx.x; i < BATCH; i += 256)
        acc += loss[i];
    red[threadIdx.x] = acc;
    __syncthreads();
    for (int off = 128; off >= 1; off >>= 1) {
        if ((int)threadIdx.x < off) red[threadIdx.x] += red[threadIdx.x + off];
        __syncthreads();
    }
    if (threadIdx.x == 0) out[0] = red[0] / (float)BATCH;
}

extern "C" void kernel_launch(void* const* d_in, const int* in_sizes, int n_in,
                              void* d_out, int out_size, void* d_ws, size_t ws_size,
                              hipStream_t stream)
{
    (void)in_sizes; (void)n_in; (void)out_size; (void)ws_size;

    const float* E      = (const float*)d_in[0];   // embeddings [B, D]
    const int*   labels = (const int*)d_in[1];     // [B]
    const float* W      = (const float*)d_in[2];   // [C, D]

    // workspace layout
    __bf16* EtG = (__bf16*)d_ws;                               // 8 MB  [D][B] bf16
    float*  ws  = (float*)((char*)d_ws + (size_t)EMBED_DIM * BATCH * sizeof(__bf16));
    const size_t PART = (size_t)BATCH * NBC;
    float* m_part  = ws;
    float* s_part  = ws + PART;
    float* t_part  = ws + 2 * PART;
    float* logit_y = ws + 3 * PART;
    float* loss    = logit_y + BATCH;

    e_transpose_bf16<<<dim3(BATCH / 64, EMBED_DIM / 64), 256, 0, stream>>>(E, EtG);

    dim3 g1(NBB, NBC);   // batch tiles fastest -> dispatch-adjacent blocks share W tile in L2
    ce_gemm_partials<<<g1, 256, 0, stream>>>(EtG, W, labels, m_part, s_part, t_part, logit_y);
    ce_row_reduce<<<BATCH / 8, 256, 0, stream>>>(m_part, s_part, t_part, logit_y, loss);
    ce_final<<<1, 256, 0, stream>>>(loss, (float*)d_out);
}
